// LSTM_2293512536834
// MI455X (gfx1250) — matrix operand order repaired and checked by reading, not compiled, not run on hardware
//
#include <hip/hip_runtime.h>

typedef __attribute__((ext_vector_type(16))) _Float16 v16h;
typedef __attribute__((ext_vector_type(8)))  float    v8f;
typedef __attribute__((ext_vector_type(4)))  float    v4f;

#define T_STEPS 500
#define NBG 16u      // batch groups of 16 rows
#define NHG 32u      // hidden groups of 16 units

// d_ws layout (bytes)
#define WPACK_OFF 0x000000u   // 2 MiB: w_hh fp16 in B-fragment layout, tile nt(0..127) x kt(0..15), 1KB each
#define HBUF_OFF  0x200000u   // 512 KiB: 16 groups * 2 buffers * 16KB (h in A-fragment layout)
#define BIAS_OFF  0x280000u   // 2048 f32: b_ih + b_hh
#define CNT_OFF   0x282000u   // 16 u32 barrier counters
#define PART_OFF  0x283000u   // 256*32 f32 output partials

// ---------------------------------------------------------------------------
// Init: repack weights to fp16 fragments, combine bias, zero h buffers/counters
// Fragment layouts (wave32, contiguous k order chosen here; A and B match):
//  B (32x16 f16): lane<16 -> col N=lane, halfs 0..15 = K k0..k0+15
//                 lane>=16 -> col N=lane-16, halfs = K k0+16..k0+31
//  A (16x32 f16): lane<16 -> row M=lane, halfs 0..15 = K k0..k0+15
//                 lane>=16 -> row M=lane-16, halfs = K k0+16..k0+31
// ---------------------------------------------------------------------------
__global__ void lstm_init(const float* __restrict__ w_hh,
                          const float* __restrict__ b_ih,
                          const float* __restrict__ b_hh,
                          unsigned char* __restrict__ ws) {
  unsigned gid = blockIdx.x * blockDim.x + threadIdx.x;   // exactly 2^20 threads
  {
    unsigned h    = gid & 15u;
    unsigned lane = (gid >> 4) & 31u;
    unsigned tile = gid >> 9;            // 0..2047
    unsigned kt   = tile & 15u;
    unsigned nt   = tile >> 4;           // gate-column tile 0..127
    unsigned N    = nt * 16u + (lane & 15u);            // gate column (row of w_hh)
    unsigned K    = kt * 32u + ((lane >> 4) & 1u) * 16u + h;  // hidden index (col of w_hh)
    ((_Float16*)(ws + WPACK_OFF))[gid] = (_Float16)w_hh[N * 512u + K];
  }
  if (gid < 262144u) ((_Float16*)(ws + HBUF_OFF))[gid] = (_Float16)0.f;
  if (gid < 2048u)   ((float*)(ws + BIAS_OFF))[gid] = b_ih[gid] + b_hh[gid];
  if (gid < 16u)     ((unsigned*)(ws + CNT_OFF))[gid] = 0u;
}

// ---------------------------------------------------------------------------
// Persistent LSTM: 512 blocks x 32 threads (1 wave each).
// block -> (bg = batch group 0..15, hg = hidden group 0..31)
// Wave computes gates for 16 batch rows x 16 hidden units (4 gate tiles),
// weights resident in 64KB LDS, h exchanged via L2 with atomic barrier.
// ---------------------------------------------------------------------------
__global__ __launch_bounds__(32) void lstm_persistent(
    const float* __restrict__ x,       // (500, 256)
    const float* __restrict__ w_ih,    // (2048)
    const float* __restrict__ w_lin,   // (512)
    unsigned char* __restrict__ ws) {
  __shared__ __align__(32) _Float16 smem[4 * 16 * 32 * 16];  // 4 gate tiles * 16KB = 64KB

  const unsigned bg   = blockIdx.x >> 5;
  const unsigned hg   = blockIdx.x & 31u;
  const unsigned lane = threadIdx.x;        // 0..31
  const unsigned n    = lane & 15u;
  const unsigned hi   = lane >> 4;
  const unsigned b0   = bg * 16u;
  const unsigned j    = hg * 16u + n;       // hidden unit owned by this lane (N in C/D tile)

  const float*   bias     = (const float*)(ws + BIAS_OFF);
  unsigned*      cnt      = (unsigned*)(ws + CNT_OFF);
  float*         partials = (float*)(ws + PART_OFF);
  unsigned char* hbase    = ws + HBUF_OFF;

  // Stage this WG's 4 gate-tile weight blocks (64KB) from L2 into LDS once.
  {
    const v4f* src = (const v4f*)(ws + WPACK_OFF);
    v4f*       dst = (v4f*)smem;
    for (unsigned idx = lane; idx < 4096u; idx += 32u) {
      unsigned gi = idx >> 10, pos = idx & 1023u;
      unsigned nt = gi * 32u + hg;               // gate tile: gi*512 + hg*16 columns
      dst[idx] = src[nt * 1024u + pos];
    }
  }
  __syncthreads();

  // Loop invariants
  float wihv[4], biasv[4];
#pragma unroll
  for (int gi = 0; gi < 4; ++gi) {
    wihv[gi]  = w_ih[(unsigned)gi * 512u + j];
    biasv[gi] = bias[(unsigned)gi * 512u + j];
  }
  // A-fragment scatter position for writing h[m][j] (m = r + 8*hi varies per r)
  const unsigned kk   = j & 31u, ktw = j >> 5;
  const unsigned b1   = (kk >> 4) & 1u;
  const unsigned hpos = kk & 15u;
  const unsigned wbyte_base = ktw * 1024u + (hi * 8u + b1 * 16u) * 32u + hpos * 2u;

  float c[8], hv[8];
#pragma unroll
  for (int r = 0; r < 8; ++r) { c[r] = 0.f; hv[r] = 0.f; }

  for (int t = 0; t < T_STEPS; ++t) {
    const unsigned rd = (unsigned)t & 1u;
    const unsigned char* hcur  = hbase + (bg * 2u + rd) * 16384u;
    unsigned char*       hnext = hbase + (bg * 2u + (rd ^ 1u)) * 16384u;

    // x contribution: gates[m][g] init = x[t][b0+m]*w_ih[g] + (b_ih+b_hh)[g]
    float x8[8];
#pragma unroll
    for (int r = 0; r < 8; ++r)
      x8[r] = x[(unsigned)t * 256u + b0 + (unsigned)r + 8u * hi];

    v8f acc[4];
#pragma unroll
    for (int gi = 0; gi < 4; ++gi)
#pragma unroll
      for (int r = 0; r < 8; ++r)
        acc[gi][r] = fmaf(x8[r], wihv[gi], biasv[gi]);

    // gates += h @ w_hh^T : K=512 as 16 k-tiles; A from L2 h buffer, B from LDS
#pragma unroll 4
    for (unsigned kt = 0; kt < 16u; ++kt) {
      v16h a = *(const v16h*)(hcur + kt * 1024u + lane * 32u);
#pragma unroll
      for (unsigned gi = 0; gi < 4u; ++gi) {
        v16h b = *(const v16h*)((const unsigned char*)smem +
                                gi * 16384u + kt * 1024u + lane * 32u);
        acc[gi] = __builtin_amdgcn_wmma_f32_16x16x32_f16(
            false, a, false, b, (short)0, acc[gi], false, false);
      }
    }

    // activations + cell/state update (pytorch order i,f,g,o)
#pragma unroll
    for (int r = 0; r < 8; ++r) {
      float ig = 1.f / (1.f + __expf(-acc[0][r]));
      float fg = 1.f / (1.f + __expf(-acc[1][r]));
      float zg = fminf(fmaxf(acc[2][r], -12.f), 12.f);
      float e2 = __expf(2.f * zg);
      float gg = (e2 - 1.f) / (e2 + 1.f);
      float og = 1.f / (1.f + __expf(-acc[3][r]));
      float cn = fg * c[r] + ig * gg;
      c[r] = cn;
      float ct  = fminf(fmaxf(cn, -12.f), 12.f);
      float e2c = __expf(2.f * ct);
      hv[r] = og * (e2c - 1.f) / (e2c + 1.f);
    }

    // publish new h slice in A-fragment layout, then group barrier
#pragma unroll
    for (int r = 0; r < 8; ++r)
      *(_Float16*)(hnext + wbyte_base + (unsigned)r * 32u) = (_Float16)hv[r];

    __builtin_amdgcn_fence(__ATOMIC_RELEASE, "agent");
    if (lane == 0)
      __hip_atomic_fetch_add(&cnt[bg], 1u, __ATOMIC_RELAXED, __HIP_MEMORY_SCOPE_AGENT);
    const unsigned target = 32u * (unsigned)(t + 1);
    while (__hip_atomic_load(&cnt[bg], __ATOMIC_RELAXED, __HIP_MEMORY_SCOPE_AGENT) < target)
      __builtin_amdgcn_s_sleep(2);
    __builtin_amdgcn_fence(__ATOMIC_ACQUIRE, "agent");
  }

  // deterministic partial of h_last @ w_lin^T over this WG's 16 hidden units
  const float wl = w_lin[j];
  __syncthreads();                       // weights no longer needed; reuse LDS
  float* red = (float*)smem;             // [32 lanes][8]
#pragma unroll
  for (int r = 0; r < 8; ++r) red[lane * 8u + (unsigned)r] = hv[r] * wl;
  __syncthreads();
  if (lane < 16u) {
    unsigned m = lane, mh = m >> 3, mr = m & 7u;
    float s = 0.f;
#pragma unroll
    for (unsigned nn = 0; nn < 16u; ++nn) s += red[(nn + 16u * mh) * 8u + mr];
    partials[(b0 + m) * 32u + hg] = s;
  }
}

// Final deterministic reduce: out[b] = b_lin + sum over 32 hidden groups
__global__ void lstm_final(const float* __restrict__ b_lin,
                           const unsigned char* __restrict__ ws,
                           float* __restrict__ out) {
  unsigned b = threadIdx.x;  // 256 threads
  const float* partials = (const float*)(ws + PART_OFF);
  float s = b_lin[0];
#pragma unroll 8
  for (unsigned hg = 0; hg < 32u; ++hg) s += partials[b * 32u + hg];
  out[b] = s;
}

extern "C" void kernel_launch(void* const* d_in, const int* in_sizes, int n_in,
                              void* d_out, int out_size, void* d_ws, size_t ws_size,
                              hipStream_t stream) {
  const float* x     = (const float*)d_in[0];
  const float* w_ih  = (const float*)d_in[1];
  const float* w_hh  = (const float*)d_in[2];
  const float* b_ih  = (const float*)d_in[3];
  const float* b_hh  = (const float*)d_in[4];
  const float* w_lin = (const float*)d_in[5];
  const float* b_lin = (const float*)d_in[6];
  unsigned char* ws  = (unsigned char*)d_ws;

  lstm_init<<<4096, 256, 0, stream>>>(w_hh, b_ih, b_hh, ws);          // 2^20 threads
  lstm_persistent<<<NBG * NHG, 32, 0, stream>>>(x, w_ih, w_lin, ws);  // 512 waves
  lstm_final<<<1, 256, 0, stream>>>(b_lin, ws, (float*)d_out);
}
